// MultiScaleG_TCN_32066225832588
// MI455X (gfx1250) — compile-verified, matrix operand order checked by reading, not run on hardware
//
#include <hip/hip_runtime.h>
#include <hip/hip_bf16.h>

// ---------------------------------------------------------------------------
// Types & WMMA helpers (gfx1250, wave32)
// ---------------------------------------------------------------------------
typedef __attribute__((ext_vector_type(16))) __bf16 v16bf;
typedef __attribute__((ext_vector_type(8)))  float  v8f;

__device__ inline unsigned short f2bf(float x){
  unsigned u = __float_as_uint(x);
  unsigned r = (u + 0x7fffu + ((u >> 16) & 1u)) >> 16;   // RNE
  return (unsigned short)r;
}
__device__ inline float bf2f(unsigned short h){
  return __uint_as_float(((unsigned)h) << 16);
}

union BFrag { v16bf v; unsigned short u[16]; };

// K-index map for 16-bit fragments (ISA 7.12.2, 16-bit A 16x32 layout)
__device__ inline int kmap(int i, int hi){
  return (i < 8) ? (hi * 8 + i) : (16 + hi * 8 + (i - 8));
}

// Row-major fragment loader: P is [idx, k] bf16 row-major with leading dim ld.
// Lane holds logical row idx0+(lane&15); elements cover k0..k0+31 (kmap order).
// Used for A tiles (idx = M row) and for transposed-B tiles (idx = N col).
// Per lane this is two contiguous 8-element (16B) runs -> vectorizable loads.
__device__ inline v16bf load_frag(const unsigned short* P, int ld, int idx0, int k0){
  int lane = threadIdx.x & 31, hi = lane >> 4;
  const unsigned short* p = P + (size_t)(idx0 + (lane & 15)) * ld + k0;
  BFrag f;
#pragma unroll
  for (int i = 0; i < 16; i++) f.u[i] = p[kmap(i, hi)];
  return f.v;
}
__device__ inline v8f wmma_bf(v16bf a, v16bf b, v8f c){
  return __builtin_amdgcn_wmma_f32_16x16x32_bf16(false, a, false, b, (short)0, c, false, false);
}

// monotonic float <-> uint encode for atomicMax-based segment max
__device__ inline unsigned fenc(float x){
  unsigned u = __float_as_uint(x);
  return (u & 0x80000000u) ? ~u : (u | 0x80000000u);
}
__device__ inline float fdec(unsigned k){
  unsigned u = (k & 0x80000000u) ? (k ^ 0x80000000u) : ~k;
  return __uint_as_float(u);
}

// ---------------------------------------------------------------------------
// Small utility kernels
// ---------------------------------------------------------------------------
__global__ void k_fill_f32(float* p, float v, int n){
  int g = blockIdx.x * blockDim.x + threadIdx.x; if (g < n) p[g] = v;
}
__global__ void k_fill_u32(unsigned* p, unsigned v, int n){
  int g = blockIdx.x * blockDim.x + threadIdx.x; if (g < n) p[g] = v;
}
__global__ void k_cvt_bf16(unsigned short* d, const float* s, int n){
  int g = blockIdx.x * blockDim.x + threadIdx.x; if (g < n) d[g] = f2bf(s[g]);
}
// pad columns: src [M,Ks] -> dst [M,Kd], zeros beyond Ks (activation K padding)
__global__ void k_cvt_pad_cols(unsigned short* d, const float* s, int M, int Ks, int Kd){
  int g = blockIdx.x * blockDim.x + threadIdx.x;
  if (g >= M * Kd) return;
  int r = g / Kd, c = g - r * Kd;
  d[g] = (c < Ks) ? f2bf(s[(size_t)r * Ks + c]) : (unsigned short)0;
}
// transpose + pad: src f32 [K,N] -> dst bf16 [Nd][Kd]; zero outside [K,N]
__global__ void k_cvt_T(unsigned short* d, const float* s, int K, int N, int Kd, int Nd){
  int g = blockIdx.x * blockDim.x + threadIdx.x;
  if (g >= Nd * Kd) return;
  int n = g / Kd, k = g - n * Kd;
  d[g] = (n < N && k < K) ? f2bf(s[(size_t)k * N + n]) : (unsigned short)0;
}
// conv weight: src [64(co),64(ci),K] -> dst [K(tap, time-reversed)][64(co)][64(ci)]
__global__ void k_cvt_convT(unsigned short* d, const float* s, int K){
  int g = blockIdx.x * blockDim.x + threadIdx.x;
  if (g >= K * 4096) return;
  int j = g >> 12, co = (g >> 6) & 63, ci = g & 63;
  d[g] = f2bf(s[(size_t)(co * 64 + ci) * K + (K - 1 - j)]);
}
__global__ void k_bn_prep(float* sc, float* sh, const float* g, const float* b,
                          const float* m, const float* v, int C){
  int i = blockIdx.x * blockDim.x + threadIdx.x;
  if (i < C){ float s = g[i] * rsqrtf(v[i] + 1e-5f); sc[i] = s; sh[i] = b[i] - m[i] * s; }
}

// ---------------------------------------------------------------------------
// GEMM (for GAT projections): C[M,N] = A[M,K] @ B[K,N], Bt stored [N][K].
// One wave computes 16 rows x 64 cols (4 accumulators, A loaded once per k).
// ---------------------------------------------------------------------------
__global__ __launch_bounds__(128) void k_gemm64(const unsigned short* A, int lda,
                                                const unsigned short* Bt, int ldb,
                                                unsigned short* C, int ldc,
                                                int M, int Ncol, int K){
  int wid = blockIdx.x * 4 + (threadIdx.x >> 5);
  int nquads = Ncol >> 6;
  int mt = wid / nquads, nq = wid - mt * nquads;
  if (mt * 16 >= M) return;
  v8f z = {0,0,0,0,0,0,0,0};
  v8f acc[4] = {z, z, z, z};
  for (int k0 = 0; k0 < K; k0 += 32){
    v16bf a = load_frag(A, lda, mt * 16, k0);
#pragma unroll
    for (int c = 0; c < 4; c++){
      v16bf b = load_frag(Bt, ldb, nq * 64 + c * 16, k0);
      acc[c] = wmma_bf(a, b, acc[c]);
    }
  }
  int lane = threadIdx.x & 31, hi = lane >> 4, lr = lane & 15;
#pragma unroll
  for (int c = 0; c < 4; c++){
    int col = nq * 64 + c * 16 + lr;
#pragma unroll
    for (int v = 0; v < 8; v++){
      int row = mt * 16 + v + 8 * hi;
      C[(size_t)row * ldc + col] = f2bf(acc[c][v]);
    }
  }
}

// ---------------------------------------------------------------------------
// Fused 2-layer MLP (K1 -> 64 -> 64, relu both). One wave handles 16 rows and
// all 64 cols; h1 staged in a per-wave LDS slice (same-wave DS order => no
// barrier). W1t [64][K1], W2t [64][64] transposed bf16.
// ---------------------------------------------------------------------------
__global__ __launch_bounds__(128) void k_mlp2(const unsigned short* A, int lda,
                                              const unsigned short* W1t, const float* b1,
                                              const unsigned short* W2t, const float* b2,
                                              unsigned short* Out, int K1, int M){
  __shared__ unsigned short h1s[4][16 * 64];
  int wave = threadIdx.x >> 5;
  int mt = blockIdx.x * 4 + wave;
  if (mt * 16 >= M) return;
  unsigned short* hs = h1s[wave];
  int lane = threadIdx.x & 31, hi = lane >> 4, lr = lane & 15;
  v8f z = {0,0,0,0,0,0,0,0};
  v8f acc[4] = {z, z, z, z};
  for (int k0 = 0; k0 < K1; k0 += 32){
    v16bf a = load_frag(A, lda, mt * 16, k0);
#pragma unroll
    for (int c = 0; c < 4; c++){
      v16bf b = load_frag(W1t, K1, c * 16, k0);
      acc[c] = wmma_bf(a, b, acc[c]);
    }
  }
#pragma unroll
  for (int c = 0; c < 4; c++){
    int col = c * 16 + lr;
    float bi = b1[col];
#pragma unroll
    for (int v = 0; v < 8; v++){
      int r = v + 8 * hi;
      float x = acc[c][v] + bi; x = x > 0.f ? x : 0.f;
      hs[r * 64 + col] = f2bf(x);
    }
  }
  v8f acc2[4] = {z, z, z, z};
  for (int k0 = 0; k0 < 64; k0 += 32){
    v16bf a = load_frag(hs, 64, 0, k0);
#pragma unroll
    for (int c = 0; c < 4; c++){
      v16bf b = load_frag(W2t, 64, c * 16, k0);
      acc2[c] = wmma_bf(a, b, acc2[c]);
    }
  }
#pragma unroll
  for (int c = 0; c < 4; c++){
    int col = c * 16 + lr;
    float bi = b2[col];
#pragma unroll
    for (int v = 0; v < 8; v++){
      int row = mt * 16 + v + 8 * hi;
      float x = acc2[c][v] + bi; x = x > 0.f ? x : 0.f;
      Out[(size_t)row * 64 + col] = f2bf(x);
    }
  }
}

// ---------------------------------------------------------------------------
// TCN causal dilated conv + bias + BN + ReLU (+optional residual + ReLU)
// X,Y layout [T][R][64] bf16; WtT [ktaps][64co][64ci] (time reversed).
// One wave per (t, row-tile): 16 rows x 64 cols, taps accumulated.
// ---------------------------------------------------------------------------
__global__ __launch_bounds__(128) void k_tcn_conv(const unsigned short* X,
                                                  const unsigned short* WtT,
                                                  const float* bias,
                                                  const float* bnsc, const float* bnsh,
                                                  const unsigned short* Res,
                                                  unsigned short* Y,
                                                  int R, int T, int ktaps, int dil){
  int wid = blockIdx.x * 4 + (threadIdx.x >> 5);
  int rtiles = R >> 4;
  int t = wid / rtiles, rt = wid - t * rtiles;
  if (t >= T) return;
  v8f z = {0,0,0,0,0,0,0,0};
  v8f acc[4] = {z, z, z, z};
  for (int j = 0; j < ktaps; j++){
    int ts = t - j * dil;
    if (ts < 0) continue;                 // zero-padded rows contribute nothing
    const unsigned short* Xt = X + ((size_t)ts * R + rt * 16) * 64;
    if (ts - dil >= 0)                    // prefetch next tap's activation tile
      __builtin_prefetch(X + ((size_t)(ts - dil) * R + rt * 16) * 64, 0, 3);
    const unsigned short* Wj = WtT + (size_t)j * 4096;
    for (int k0 = 0; k0 < 64; k0 += 32){
      v16bf a = load_frag(Xt, 64, 0, k0);
#pragma unroll
      for (int c = 0; c < 4; c++){
        v16bf b = load_frag(Wj, 64, c * 16, k0);
        acc[c] = wmma_bf(a, b, acc[c]);
      }
    }
  }
  int lane = threadIdx.x & 31, hi = lane >> 4, lr = lane & 15;
#pragma unroll
  for (int c = 0; c < 4; c++){
    int col = c * 16 + lr;
    float sc = bnsc[col], sh = bnsh[col], bi = bias[col];
#pragma unroll
    for (int v = 0; v < 8; v++){
      int r = rt * 16 + v + 8 * hi;
      size_t idx = ((size_t)t * R + r) * 64 + col;
      float x = (acc[c][v] + bi) * sc + sh;
      x = x > 0.f ? x : 0.f;
      if (Res){ x += bf2f(Res[idx]); x = x > 0.f ? x : 0.f; }
      Y[idx] = f2bf(x);
    }
  }
}

// ---------------------------------------------------------------------------
// GAT helper kernels (f32 epilogues)
// ---------------------------------------------------------------------------
__global__ void k_row_mean(unsigned short* mean, const unsigned short* X, int R){
  __shared__ float part[256];
  int c = threadIdx.x & 63, p = threadIdx.x >> 6;
  float s = 0.f;
  for (int r = p; r < R; r += 4) s += bf2f(X[(size_t)r * 64 + c]);
  part[threadIdx.x] = s;
  __syncthreads();
  if (p == 0){
    float tot = part[c] + part[64 + c] + part[128 + c] + part[192 + c];
    mean[c] = f2bf(tot / (float)R);
  }
}
// el_loop[o] = mean(64) @ w_edge[64,256]; Wt stored [256][64]
__global__ void k_vecmat256(unsigned short* out, const unsigned short* mean,
                            const unsigned short* Wt){
  int o = threadIdx.x; float s = 0.f;
  for (int i = 0; i < 64; i++) s += bf2f(mean[i]) * bf2f(Wt[o * 64 + i]);
  out[o] = f2bf(s);
}
__global__ void k_aeloop(float* out, const unsigned short* elp, const float* aedge){
  int h = threadIdx.x;
  if (h < 4){
    float s = 0.f;
    for (int c = 0; c < 64; c++) s += bf2f(elp[h * 64 + c]) * aedge[h * 64 + c];
    out[h] = s;
  }
}
__global__ void k_att_node(float* as_, float* ad_, const unsigned short* xl,
                           const float* asrc, const float* adst, int Nn){
  int g = blockIdx.x * blockDim.x + threadIdx.x;
  if (g >= Nn * 4) return;
  int n = g >> 2, h = g & 3;
  const unsigned short* x = xl + (size_t)n * 256 + h * 64;
  float s = 0.f, d = 0.f;
  for (int c = 0; c < 64; c++){
    float v = bf2f(x[c]); s += v * asrc[h * 64 + c]; d += v * adst[h * 64 + c];
  }
  as_[g] = s; ad_[g] = d;
}
__global__ void k_att_edge(float* ae_, const unsigned short* el, const float* aedge, int E){
  int g = blockIdx.x * blockDim.x + threadIdx.x;
  if (g >= E * 4) return;
  int e = g >> 2, h = g & 3;
  const unsigned short* x = el + (size_t)e * 256 + h * 64;
  float s = 0.f;
  for (int c = 0; c < 64; c++) s += bf2f(x[c]) * aedge[h * 64 + c];
  ae_[g] = s;
}
__global__ void k_att_logits(float* alog, unsigned* mkey, const float* as_, const float* ad_,
                             const float* ae_, const float* aeloop, const int* ei,
                             int E, int Nn){
  int g = blockIdx.x * blockDim.x + threadIdx.x;
  if (g >= (E + Nn) * 4) return;
  int e = g >> 2, h = g & 3;
  int s, d; float av;
  if (e < E){ s = ei[e]; d = ei[E + e]; av = ae_[g]; }
  else      { s = d = e - E; av = aeloop[h]; }
  float a = as_[s * 4 + h] + ad_[d * 4 + h] + av;
  a = a > 0.f ? a : 0.2f * a;           // leaky relu
  alog[g] = a;
  atomicMax(&mkey[d * 4 + h], fenc(a));
}
__global__ void k_att_exp(float* expw, float* ssum, const float* alog, const unsigned* mkey,
                          const int* ei, int E, int Nn){
  int g = blockIdx.x * blockDim.x + threadIdx.x;
  if (g >= (E + Nn) * 4) return;
  int e = g >> 2, h = g & 3;
  int d = (e < E) ? ei[E + e] : (e - E);
  float w = __expf(alog[g] - fdec(mkey[d * 4 + h]));
  expw[g] = w;
  atomicAdd(&ssum[d * 4 + h], w);
}
__global__ void k_gat_agg(float* agg, const float* expw, const float* ssum,
                          const unsigned short* xl, const int* ei, int E, int Nn){
  int g = blockIdx.x * blockDim.x + threadIdx.x;
  if (g >= (E + Nn) * 256) return;
  int e = g >> 8, r = g & 255, h = r >> 6;
  int s, d;
  if (e < E){ s = ei[e]; d = ei[E + e]; } else { s = d = e - E; }
  float w = expw[e * 4 + h] / ssum[d * 4 + h];
  atomicAdd(&agg[(size_t)d * 256 + r], w * bf2f(xl[(size_t)s * 256 + r]));
}
__global__ void k_gat_finish(unsigned short* out, const float* agg, const float* gbias,
                             const float* bnsc, const float* bnsh, int Nn){
  int g = blockIdx.x * blockDim.x + threadIdx.x;
  if (g >= Nn * 64) return;
  int n = g >> 6, c = g & 63;
  const float* a = agg + (size_t)n * 256;
  float v = 0.25f * (a[c] + a[64 + c] + a[128 + c] + a[192 + c]) + gbias[c];
  v = v * bnsc[c] + bnsh[c];
  v = v > 0.f ? v : 0.f;
  out[g] = f2bf(v);
}

// ---------------------------------------------------------------------------
// Fused classifier: feats[576] = [tn@src | tn@dst | te] gathered per row,
// clf1(576->128)+relu staged in LDS, clf2(128->8) by wave 0.
// 64 threads (2 waves) per 16-row tile; each wave computes 16x64 of h.
// W1t [128][576], W2t [16][128] (rows 8..15 zero).
// ---------------------------------------------------------------------------
__global__ __launch_bounds__(64) void k_clf(const unsigned short* nb0, const unsigned short* nb1,
                                            const unsigned short* nb2, const unsigned short* eb0,
                                            const unsigned short* eb1, const unsigned short* eb2,
                                            const unsigned short* W1t, const float* B1,
                                            const unsigned short* W2t, const float* B2,
                                            const int* ei, int E, int Nn, float* out){
  __shared__ int sS[16], sD[16], sE[16];
  __shared__ unsigned short h[16 * 128];
  int mt = blockIdx.x;
  int t = (mt * 16) / E;
  int ebase = mt * 16 - t * E;
  if (threadIdx.x < 16){
    int e = ebase + threadIdx.x;
    sE[threadIdx.x] = e;
    sS[threadIdx.x] = ei[e];
    sD[threadIdx.x] = ei[E + e];
  }
  __syncthreads();
  int wave = threadIdx.x >> 5, lane = threadIdx.x & 31, hi = lane >> 4, lr = lane & 15;
  v8f z = {0,0,0,0,0,0,0,0};
  v8f acc[4] = {z, z, z, z};
  for (int k0 = 0; k0 < 576; k0 += 32){
    BFrag fa;
#pragma unroll
    for (int i = 0; i < 16; i++){
      int k = k0 + kmap(i, hi);
      int reg = k >> 6, c = k & 63;
      int row; size_t Rd; const unsigned short* bp;
      if (reg < 3){ row = sS[lr]; Rd = (size_t)Nn; bp = (reg == 0) ? nb0 : (reg == 1) ? nb1 : nb2; }
      else if (reg < 6){ row = sD[lr]; Rd = (size_t)Nn; bp = (reg == 3) ? nb0 : (reg == 4) ? nb1 : nb2; }
      else { row = sE[lr]; Rd = (size_t)E; bp = (reg == 6) ? eb0 : (reg == 7) ? eb1 : eb2; }
      fa.u[i] = bp[((size_t)t * Rd + row) * 64 + c];
    }
#pragma unroll
    for (int c = 0; c < 4; c++){
      v16bf b = load_frag(W1t, 576, wave * 64 + c * 16, k0);
      acc[c] = wmma_bf(fa.v, b, acc[c]);
    }
  }
#pragma unroll
  for (int c = 0; c < 4; c++){
    int col = wave * 64 + c * 16 + lr;
    float bi = B1[col];
#pragma unroll
    for (int v = 0; v < 8; v++){
      int r = v + 8 * hi;
      float x = acc[c][v] + bi; x = x > 0.f ? x : 0.f;
      h[r * 128 + col] = f2bf(x);
    }
  }
  __syncthreads();
  if (wave == 0){
    v8f a2 = z;
    for (int k0 = 0; k0 < 128; k0 += 32){
      v16bf a = load_frag(h, 128, 0, k0);
      v16bf b = load_frag(W2t, 128, 0, k0);
      a2 = wmma_bf(a, b, a2);
    }
    if (lr < 8){
      float bi = B2[lr];
#pragma unroll
      for (int v = 0; v < 8; v++){
        size_t m = (size_t)mt * 16 + v + 8 * hi;
        out[m * 8 + lr] = a2[v] + bi;
      }
    }
  }
}

// ---------------------------------------------------------------------------
// Host orchestration
// ---------------------------------------------------------------------------
static inline int cdiv(int a, int b){ return (a + b - 1) / b; }

extern "C" void kernel_launch(void* const* d_in, const int* in_sizes, int n_in,
                              void* d_out, int out_size, void* d_ws, size_t ws_size,
                              hipStream_t stream){
  (void)in_sizes; (void)n_in; (void)out_size; (void)ws_size;
  const int T = 8, Nn = 4000, E = 48000, ESL = E + Nn;
  const int KS[3] = {3, 5, 7};

  // ---- parse inputs (setup_inputs insertion order, params pytree in insertion order)
  int ip = 0;
  auto F = [&](){ return (const float*)d_in[ip++]; };
  const float* x_seq  = F();                       // [T,N,32]
  const float* ea_seq = F();                       // [T,E,16]
  const int*   eidx   = (const int*)d_in[ip++];    // [2,E]
  ip++;                                            // seq_len
  const float *nfc1w=F(),*nfc1b=F(),*nfc2w=F(),*nfc2b=F();
  const float *efc1iw=F(),*efc1ib=F(),*efc2iw=F(),*efc2ib=F();
  const float *g1wl=F(),*g1we=F(),*g1as=F(),*g1ad=F(),*g1ae=F(),*g1b=F();
  const float *g2wl=F(),*g2we=F(),*g2as=F(),*g2ad=F(),*g2ae=F(),*g2b=F();
  const float *bn1g=F(),*bn1b=F(),*bn1m=F(),*bn1v=F();
  const float *bn2g=F(),*bn2b=F(),*bn2m=F(),*bn2v=F();
  const float *efc1w=F(),*efc1b=F(),*efc2w=F(),*efc2b=F();
  const float* tnp[3][2][12];
  for (int br = 0; br < 3; br++) for (int bl = 0; bl < 2; bl++)
    for (int f = 0; f < 12; f++) tnp[br][bl][f] = F();
  const float* tep[3][2][12];
  for (int br = 0; br < 3; br++) for (int bl = 0; bl < 2; bl++)
    for (int f = 0; f < 12; f++) tep[br][bl][f] = F();
  const float *c1w=F(),*c1b=F(),*c2w=F(),*c2b=F();

  // ---- workspace bump allocator
  char* wsp = (char*)d_ws; size_t off = 0;
  auto alloc = [&](size_t bytes)->void*{
    off = (off + 255) & ~(size_t)255;
    void* p = wsp + off; off += bytes; return p;
  };
  auto aU = [&](size_t n){ return (unsigned short*)alloc(n * 2); };
  auto aF = [&](size_t n){ return (float*)alloc(n * 4); };

  unsigned short* xin_bf = aU((size_t)T*Nn*32);
  unsigned short* ein_bf = aU((size_t)T*E*32);
  // All weights stored TRANSPOSED: [out][in] bf16 (K-padded where needed)
  unsigned short* w_nfc1 = aU(64*32);
  unsigned short* w_nfc2 = aU(64*64);
  unsigned short* w_efc1i= aU(64*32);          // K padded 16->32
  unsigned short* w_efc2i= aU(64*64);
  unsigned short* w_efc1 = aU(64*64);
  unsigned short* w_efc2 = aU(64*64);
  unsigned short* w_g1l  = aU(256*64);
  unsigned short* w_g1e  = aU(256*64);
  unsigned short* w_g2l  = aU(256*64);
  unsigned short* w_g2e  = aU(256*64);
  unsigned short* w_clf1 = aU(128*576);
  unsigned short* w_clf2 = aU(16*128);         // N padded 8->16
  unsigned short* tnw[3][2][2]; unsigned short* tew[3][2][2];
  float *tnsc[3][2][2], *tnsh[3][2][2], *tesc[3][2][2], *tesh[3][2][2];
  for (int br = 0; br < 3; br++) for (int bl = 0; bl < 2; bl++) for (int cv = 0; cv < 2; cv++){
    tnw[br][bl][cv] = aU((size_t)KS[br]*4096);
    tew[br][bl][cv] = aU((size_t)KS[br]*4096);
    tnsc[br][bl][cv] = aF(64); tnsh[br][bl][cv] = aF(64);
    tesc[br][bl][cv] = aF(64); tesh[br][bl][cv] = aF(64);
  }
  float *bnn1sc=aF(64), *bnn1sh=aF(64), *bnn2sc=aF(64), *bnn2sh=aF(64);

  unsigned short* xh   = aU((size_t)T*Nn*64);    // node MLP out
  unsigned short* ea   = aU((size_t)T*E*64);     // edge MLP-in out
  unsigned short* e2   = aU((size_t)T*E*64);     // edge_fc out
  unsigned short* nseq = aU((size_t)T*Nn*64);    // GAT layer2 out
  unsigned short* xB   = aU((size_t)Nn*64);
  unsigned short* xl   = aU((size_t)Nn*256);
  unsigned short* el   = aU((size_t)E*256);
  unsigned short* meanv= aU(64);
  unsigned short* elloop = aU(256);
  float* aeloop = aF(4);
  float* as_ = aF((size_t)Nn*4);  float* ad_ = aF((size_t)Nn*4);
  float* aeE = aF((size_t)E*4);
  float* alog = aF((size_t)ESL*4); float* expw = aF((size_t)ESL*4);
  unsigned* mkey = (unsigned*)alloc((size_t)Nn*4*4);
  float* ssum = aF((size_t)Nn*4);
  float* agg  = aF((size_t)Nn*256);
  unsigned short* ntA = aU((size_t)T*Nn*64);
  unsigned short* ntB = aU((size_t)T*Nn*64);
  unsigned short* nb[3]; for (int i = 0; i < 3; i++) nb[i] = aU((size_t)T*Nn*64);
  unsigned short* etA = aU((size_t)T*E*64);
  unsigned short* etB = aU((size_t)T*E*64);
  unsigned short* eb[3]; for (int i = 0; i < 3; i++) eb[i] = aU((size_t)T*E*64);

  auto G = [](int n){ return cdiv(n, 256); };

  // ---- input conversion (activations stay row-major [row, k])
  k_cvt_bf16<<<G(T*Nn*32),256,0,stream>>>(xin_bf, x_seq, T*Nn*32);
  k_cvt_pad_cols<<<G(T*E*32),256,0,stream>>>(ein_bf, ea_seq, T*E, 16, 32);
  // ---- weight conversion: transpose to [N][K] bf16
  k_cvt_T<<<G(64*32),256,0,stream>>>(w_nfc1, nfc1w, 32, 64, 32, 64);
  k_cvt_T<<<G(64*64),256,0,stream>>>(w_nfc2, nfc2w, 64, 64, 64, 64);
  k_cvt_T<<<G(64*32),256,0,stream>>>(w_efc1i, efc1iw, 16, 64, 32, 64);
  k_cvt_T<<<G(64*64),256,0,stream>>>(w_efc2i, efc2iw, 64, 64, 64, 64);
  k_cvt_T<<<G(64*64),256,0,stream>>>(w_efc1, efc1w, 64, 64, 64, 64);
  k_cvt_T<<<G(64*64),256,0,stream>>>(w_efc2, efc2w, 64, 64, 64, 64);
  k_cvt_T<<<G(256*64),256,0,stream>>>(w_g1l, g1wl, 64, 256, 64, 256);
  k_cvt_T<<<G(256*64),256,0,stream>>>(w_g1e, g1we, 64, 256, 64, 256);
  k_cvt_T<<<G(256*64),256,0,stream>>>(w_g2l, g2wl, 64, 256, 64, 256);
  k_cvt_T<<<G(256*64),256,0,stream>>>(w_g2e, g2we, 64, 256, 64, 256);
  k_cvt_T<<<G(128*576),256,0,stream>>>(w_clf1, c1w, 576, 128, 576, 128);
  k_cvt_T<<<G(16*128),256,0,stream>>>(w_clf2, c2w, 128, 8, 128, 16);
  k_bn_prep<<<1,64,0,stream>>>(bnn1sc, bnn1sh, bn1g, bn1b, bn1m, bn1v, 64);
  k_bn_prep<<<1,64,0,stream>>>(bnn2sc, bnn2sh, bn2g, bn2b, bn2m, bn2v, 64);
  for (int br = 0; br < 3; br++) for (int bl = 0; bl < 2; bl++) for (int cv = 0; cv < 2; cv++){
    int wf = cv ? 6 : 0, gf = cv ? 8 : 2;
    k_cvt_convT<<<G(KS[br]*4096),256,0,stream>>>(tnw[br][bl][cv], tnp[br][bl][wf], KS[br]);
    k_cvt_convT<<<G(KS[br]*4096),256,0,stream>>>(tew[br][bl][cv], tep[br][bl][wf], KS[br]);
    k_bn_prep<<<1,64,0,stream>>>(tnsc[br][bl][cv], tnsh[br][bl][cv],
        tnp[br][bl][gf], tnp[br][bl][gf+1], tnp[br][bl][gf+2], tnp[br][bl][gf+3], 64);
    k_bn_prep<<<1,64,0,stream>>>(tesc[br][bl][cv], tesh[br][bl][cv],
        tep[br][bl][gf], tep[br][bl][gf+1], tep[br][bl][gf+2], tep[br][bl][gf+3], 64);
  }

  // ---- input MLPs (WMMA, fused 2-layer)
  k_mlp2<<<cdiv(T*Nn/16,4),128,0,stream>>>(xin_bf, 32, w_nfc1, nfc1b, w_nfc2, nfc2b, xh, 32, T*Nn);
  k_mlp2<<<cdiv(T*E/16,4),128,0,stream>>>(ein_bf, 32, w_efc1i, efc1ib, w_efc2i, efc2ib, ea, 32, T*E);
  k_mlp2<<<cdiv(T*E/16,4),128,0,stream>>>(ea, 64, w_efc1, efc1b, w_efc2, efc2b, e2, 64, T*E);

  auto gemm = [&](const unsigned short* A, int lda, const unsigned short* Bt, int ldb,
                  unsigned short* C, int ldc, int M, int Ncol, int K){
    int waves = (M / 16) * (Ncol / 64);
    k_gemm64<<<cdiv(waves,4),128,0,stream>>>(A, lda, Bt, ldb, C, ldc, M, Ncol, K);
  };

  // ---- GAT per timestep (2 layers)
  auto gat_layer = [&](const unsigned short* xin, const unsigned short* wlin,
                       const unsigned short* wedge, const float* asrc, const float* adst,
                       const float* aedge, const float* gbias,
                       const float* bnsc, const float* bnsh,
                       const unsigned short* ea_t, unsigned short* xout){
    gemm(xin, 64, wlin, 64, xl, 256, Nn, 256, 64);
    gemm(ea_t, 64, wedge, 64, el, 256, E, 256, 64);
    k_vecmat256<<<1,256,0,stream>>>(elloop, meanv, wedge);
    k_aeloop<<<1,32,0,stream>>>(aeloop, elloop, aedge);
    k_att_node<<<G(Nn*4),256,0,stream>>>(as_, ad_, xl, asrc, adst, Nn);
    k_att_edge<<<G(E*4),256,0,stream>>>(aeE, el, aedge, E);
    k_fill_u32<<<G(Nn*4),256,0,stream>>>(mkey, 0u, Nn*4);
    k_fill_f32<<<G(Nn*4),256,0,stream>>>(ssum, 0.f, Nn*4);
    k_fill_f32<<<G(Nn*256),256,0,stream>>>(agg, 0.f, Nn*256);
    k_att_logits<<<G(ESL*4),256,0,stream>>>(alog, mkey, as_, ad_, aeE, aeloop, eidx, E, Nn);
    k_att_exp<<<G(ESL*4),256,0,stream>>>(expw, ssum, alog, mkey, eidx, E, Nn);
    k_gat_agg<<<G(ESL*256),256,0,stream>>>(agg, expw, ssum, xl, eidx, E, Nn);
    k_gat_finish<<<G(Nn*64),256,0,stream>>>(xout, agg, gbias, bnsc, bnsh, Nn);
  };

  for (int t = 0; t < T; t++){
    const unsigned short* ea_t = ea + (size_t)t * E * 64;
    k_row_mean<<<1,256,0,stream>>>(meanv, ea_t, E);
    gat_layer(xh + (size_t)t*Nn*64, w_g1l, w_g1e, g1as, g1ad, g1ae, g1b,
              bnn1sc, bnn1sh, ea_t, xB);
    gat_layer(xB, w_g2l, w_g2e, g2as, g2ad, g2ae, g2b,
              bnn2sc, bnn2sh, ea_t, nseq + (size_t)t*Nn*64);
  }

  // ---- TCNs (WMMA tap-accumulated GEMMs, BN/ReLU/residual fused)
  auto conv = [&](const unsigned short* X, const unsigned short* Wt, const float* bias,
                  const float* sc, const float* sh, const unsigned short* Res,
                  unsigned short* Y, int R, int kt, int dil){
    k_tcn_conv<<<cdiv(T*(R/16),4),128,0,stream>>>(X, Wt, bias, sc, sh, Res, Y, R, T, kt, dil);
  };
  for (int br = 0; br < 3; br++){
    int k = KS[br];
    conv(nseq, tnw[br][0][0], tnp[br][0][1], tnsc[br][0][0], tnsh[br][0][0], nullptr, ntA, Nn, k, 1);
    conv(ntA,  tnw[br][0][1], tnp[br][0][7], tnsc[br][0][1], tnsh[br][0][1], nseq,    ntB, Nn, k, 1);
    conv(ntB,  tnw[br][1][0], tnp[br][1][1], tnsc[br][1][0], tnsh[br][1][0], nullptr, ntA, Nn, k, 2);
    conv(ntA,  tnw[br][1][1], tnp[br][1][7], tnsc[br][1][1], tnsh[br][1][1], ntB,     nb[br], Nn, k, 2);
  }
  for (int br = 0; br < 3; br++){
    int k = KS[br];
    conv(e2,  tew[br][0][0], tep[br][0][1], tesc[br][0][0], tesh[br][0][0], nullptr, etA, E, k, 1);
    conv(etA, tew[br][0][1], tep[br][0][7], tesc[br][0][1], tesh[br][0][1], e2,      etB, E, k, 1);
    conv(etB, tew[br][1][0], tep[br][1][1], tesc[br][1][0], tesh[br][1][0], nullptr, etA, E, k, 2);
    conv(etA, tew[br][1][1], tep[br][1][7], tesc[br][1][1], tesh[br][1][1], etB,     eb[br], E, k, 2);
  }

  // ---- fused classifier -> d_out [T,E,8] f32
  k_clf<<<T*E/16,64,0,stream>>>(nb[0], nb[1], nb[2], eb[0], eb[1], eb[2],
                                w_clf1, c1b, w_clf2, c2b, eidx, E, Nn, (float*)d_out);
}
